// SelfOptimizingSolver_33681133535773
// MI455X (gfx1250) — matrix operand order repaired and checked by reading, not compiled, not run on hardware
//
#include <hip/hip_runtime.h>
#include <hip/hip_bf16.h>

// ---------------------------------------------------------------------------
// Types for CDNA5 WMMA
// ---------------------------------------------------------------------------
typedef __attribute__((ext_vector_type(16))) __bf16 v16bf;
typedef __attribute__((ext_vector_type(8)))  float  v8f;

struct alignas(16) u16x8 { unsigned int w[4]; };   // POD 16-byte chunk (8 bf16)
union Frag { u16x8 u[2]; v16bf v; };               // 32 bytes = one 16-bit WMMA operand

__device__ __forceinline__ unsigned short f2bf(float f) {
    unsigned u = __float_as_uint(f);
    unsigned r = (u + 0x7FFFu + ((u >> 16) & 1u)) >> 16;   // round-to-nearest-even
    return (unsigned short)r;
}

__device__ __forceinline__ float gelu_f(float x) {
    return 0.5f * x * (1.0f + erff(x * 0.70710678118654752440f));
}

// Generic pointer to __shared__ -> 32-bit LDS offset (low 32 bits of the address)
__device__ __forceinline__ unsigned lds_addr(const void* p) {
    return (unsigned)(unsigned long long)(uintptr_t)p;
}

// Async global -> LDS 16-byte copy (CDNA5 GLOBAL_LOAD_ASYNC_TO_LDS_B128, ASYNCcnt)
__device__ __forceinline__ void async_copy16(unsigned lds_off, const void* sbase, int voff_bytes) {
    asm volatile("global_load_async_to_lds_b128 %0, %1, %2"
                 :: "v"(lds_off), "v"(voff_bytes), "s"(sbase)
                 : "memory");
}
__device__ __forceinline__ void wait_async0() {
    asm volatile("s_wait_asynccnt 0x0" ::: "memory");
}

// A-operand fragment: 16-bit A 16x32 layout.
// lane<16: K in [0..7] (v0..3) and [16..23] (v4..7); lane>=16: +8.
__device__ __forceinline__ v16bf load_fragA(const unsigned short* rowptr, int g) {
    Frag t;
    t.u[0] = *(const u16x8*)(rowptr + 8 * g);
    t.u[1] = *(const u16x8*)(rowptr + 16 + 8 * g);
    return t.v;
}
// B-operand fragment: lane holds one column (N = lane&15); K same as A: [0..7],[16..23] for lane<16; +8 for lane>=16.
__device__ __forceinline__ v16bf load_fragB(const unsigned short* rowptr, int g) {
    Frag t;
    t.u[0] = *(const u16x8*)(rowptr + 8 * g);
    t.u[1] = *(const u16x8*)(rowptr + 16 + 8 * g);
    return t.v;
}

#define WMMA_BF16(a, b, c) \
    __builtin_amdgcn_wmma_f32_16x16x32_bf16(false, (a), false, (b), (short)0, (c), false, false)

// ---------------------------------------------------------------------------
// Model constants
// ---------------------------------------------------------------------------
#define BB   256
#define SS   64
#define DD   1024
#define HH   16
#define VV   64
#define DFF  4096
#define LL   8
#define TOK  (BB * SS)          // 16384 rows
#define SCALE_ATTN 0.125f       // 64^-0.5

// ---------------------------------------------------------------------------
// fp32 -> bf16 bulk conversion (8 elems/thread)
// ---------------------------------------------------------------------------
__global__ __launch_bounds__(256) void cvt_bf16_kernel(
    const float* __restrict__ src, unsigned short* __restrict__ dst)
{
    size_t i = ((size_t)blockIdx.x * 256 + threadIdx.x) * 8;
    const float4* s = (const float4*)(src + i);
    float4 f0 = s[0], f1 = s[1];
    alignas(16) unsigned short t[8] = {
        f2bf(f0.x), f2bf(f0.y), f2bf(f0.z), f2bf(f0.w),
        f2bf(f1.x), f2bf(f1.y), f2bf(f1.z), f2bf(f1.w) };
    *(u16x8*)(dst + i) = *(const u16x8*)t;
}

// ---------------------------------------------------------------------------
// GEMM: Out[M,N] = A[M,K] @ W[N,K]^T + bias[N].  A, W pre-converted bf16.
// Block tile 64x64, 8 waves (16 rows x 32 cols each), K-step 64,
// double-buffered LDS filled by async global->LDS copies.
// ---------------------------------------------------------------------------
template<bool OUT_BF16, bool GELU_EP>
__global__ __launch_bounds__(256) void gemm_wmma_async(
    const unsigned short* __restrict__ A, const unsigned short* __restrict__ Bw,
    const float* __restrict__ bias, void* __restrict__ Outp,
    int K, int N)
{
    __shared__ __align__(16) unsigned short As[2][64][72];
    __shared__ __align__(16) unsigned short Bs[2][64][72];

    const int tid  = threadIdx.x;
    const int wave = tid >> 5;
    const int lane = tid & 31;
    const int g    = lane >> 4;
    const int r    = lane & 15;
    const int mt   = wave & 3;            // 16-row tile in block
    const int nb   = (wave >> 2) * 32;    // 32-col half in block

    const size_t mbase = (size_t)blockIdx.y * 64;
    const size_t nbase = (size_t)blockIdx.x * 64;
    const int lrow = tid >> 2;            // 0..63
    const int lkv  = (tid & 3) * 16;      // 0,16,32,48 (elems within 64-wide K step)

    const size_t arow = (mbase + lrow) * (size_t)K + lkv;   // elems
    const size_t brow = (nbase + lrow) * (size_t)K + lkv;

    v8f acc0 = {};
    v8f acc1 = {};
    const int iters = K >> 6;

    // prologue: fill buffer 0
    {
        int aoff = (int)(arow * 2);
        async_copy16(lds_addr(&As[0][lrow][lkv]),     A,  aoff);
        async_copy16(lds_addr(&As[0][lrow][lkv + 8]), A,  aoff + 16);
        int boff = (int)(brow * 2);
        async_copy16(lds_addr(&Bs[0][lrow][lkv]),     Bw, boff);
        async_copy16(lds_addr(&Bs[0][lrow][lkv + 8]), Bw, boff + 16);
    }

    int cur = 0;
    for (int it = 0; it < iters; ++it) {
        wait_async0();          // my async writes into buf 'cur' have landed
        __syncthreads();        // everyone's have; all reads of buf cur^1 done last iter

        if (it + 1 < iters) {   // prefetch next K-slab into the other buffer
            const int nxt = cur ^ 1;
            const size_t k0 = (size_t)(it + 1) << 6;
            int aoff = (int)((arow + k0) * 2);
            async_copy16(lds_addr(&As[nxt][lrow][lkv]),     A,  aoff);
            async_copy16(lds_addr(&As[nxt][lrow][lkv + 8]), A,  aoff + 16);
            int boff = (int)((brow + k0) * 2);
            async_copy16(lds_addr(&Bs[nxt][lrow][lkv]),     Bw, boff);
            async_copy16(lds_addr(&Bs[nxt][lrow][lkv + 8]), Bw, boff + 16);
        }

#pragma unroll
        for (int kk = 0; kk < 64; kk += 32) {
            v16bf a  = load_fragA(&As[cur][mt * 16 + r][kk], g);
            v16bf b0 = load_fragB(&Bs[cur][nb + r][kk], g);
            v16bf b1 = load_fragB(&Bs[cur][nb + 16 + r][kk], g);
            acc0 = WMMA_BF16(a, b0, acc0);
            acc1 = WMMA_BF16(a, b1, acc1);
        }
        cur ^= 1;
    }

    const int n0 = (int)nbase + nb + r;
    const int n1 = n0 + 16;
    const float bia0 = bias[n0];
    const float bia1 = bias[n1];
#pragma unroll
    for (int j = 0; j < 8; j++) {
        size_t row = mbase + mt * 16 + j + 8 * g;
        float v0 = acc0[j] + bia0;
        float v1 = acc1[j] + bia1;
        if (GELU_EP) { v0 = gelu_f(v0); v1 = gelu_f(v1); }
        if (OUT_BF16) {
            unsigned short* o = (unsigned short*)Outp + row * (size_t)N;
            o[n0] = f2bf(v0);
            o[n1] = f2bf(v1);
        } else {
            float* o = (float*)Outp + row * (size_t)N;
            o[n0] = v0;
            o[n1] = v1;
        }
    }
}

// ---------------------------------------------------------------------------
// Fused attention: one block per (b,h). qkv is bf16 [TOK, 3D].
// scores = q k^T * SCALE + rel_bias[s-t+63]; softmax; probs @ v.
// ---------------------------------------------------------------------------
__global__ __launch_bounds__(256) void attn_kernel(
    const unsigned short* __restrict__ qkv,
    const float* __restrict__ rel,            // 127 floats for this layer
    unsigned short* __restrict__ attnout)
{
    __shared__ __align__(16) unsigned short qs[64][72];
    __shared__ __align__(16) unsigned short ks[64][72];
    __shared__ __align__(16) unsigned short vt[64][72];   // transposed: [d][t]
    __shared__ __align__(16) unsigned short ps[64][72];   // probs bf16
    __shared__ float sc[64][68];                          // scores fp32

    const int bh = blockIdx.x;
    const int b  = bh >> 4;
    const int hh = bh & 15;
    const int tid  = threadIdx.x;
    const int wave = tid >> 5;
    const int lane = tid & 31;
    const int g = lane >> 4;
    const int r = lane & 15;

    // phase 1: q,k via async global->LDS; v via regs (transposed store)
    {
        int s  = tid >> 2;
        int d0 = (tid & 3) * 16;
        size_t base = ((size_t)b * 64 + s) * (3 * DD) + hh * 64 + d0;   // elems
        int off = (int)(base * 2);
        async_copy16(lds_addr(&qs[s][d0]),     qkv, off);
        async_copy16(lds_addr(&qs[s][d0 + 8]), qkv, off + 16);
        async_copy16(lds_addr(&ks[s][d0]),     qkv, off + DD * 2);
        async_copy16(lds_addr(&ks[s][d0 + 8]), qkv, off + DD * 2 + 16);
        alignas(16) unsigned short tmp[16];
        *(u16x8*)&tmp[0] = *(const u16x8*)(qkv + base + 2 * DD);
        *(u16x8*)&tmp[8] = *(const u16x8*)(qkv + base + 2 * DD + 8);
#pragma unroll
        for (int j = 0; j < 16; j++) vt[d0 + j][s] = tmp[j];
        wait_async0();
    }
    __syncthreads();

    // phase 2: scores = q k^T (two 16x16 tiles per wave)
#pragma unroll
    for (int t2 = 0; t2 < 2; t2++) {
        int T  = wave + t2 * 8;
        int si = T >> 2, ti = T & 3;
        v8f acc = {};
#pragma unroll
        for (int kk = 0; kk < 64; kk += 32) {
            v16bf a  = load_fragA(&qs[si * 16 + r][kk], g);
            v16bf bb = load_fragB(&ks[ti * 16 + r][kk], g);
            acc = WMMA_BF16(a, bb, acc);
        }
#pragma unroll
        for (int j = 0; j < 8; j++) {
            int s = si * 16 + j + 8 * g;
            int t = ti * 16 + r;
            sc[s][t] = acc[j] * SCALE_ATTN + rel[s - t + 63];
        }
    }
    __syncthreads();

    // phase 3: softmax rows (4 lanes per row, shuffle reduce), write probs bf16
    {
        int s = tid >> 2;
        int c = (tid & 3) * 16;
        float v[16];
        float m = -1e30f;
#pragma unroll
        for (int i = 0; i < 16; i++) { v[i] = sc[s][c + i]; m = fmaxf(m, v[i]); }
        m = fmaxf(m, __shfl_xor(m, 1, 32));
        m = fmaxf(m, __shfl_xor(m, 2, 32));
        float sum = 0.f;
#pragma unroll
        for (int i = 0; i < 16; i++) { v[i] = __expf(v[i] - m); sum += v[i]; }
        sum += __shfl_xor(sum, 1, 32);
        sum += __shfl_xor(sum, 2, 32);
        float inv = 1.0f / sum;
        alignas(16) unsigned short tmp[16];
#pragma unroll
        for (int i = 0; i < 16; i++) tmp[i] = f2bf(v[i] * inv);
        *(u16x8*)&ps[s][c]     = *(u16x8*)&tmp[0];
        *(u16x8*)&ps[s][c + 8] = *(u16x8*)&tmp[8];
    }
    __syncthreads();

    // phase 4: attn = probs @ v  (B-operand from transposed v: K=t contiguous)
#pragma unroll
    for (int t2 = 0; t2 < 2; t2++) {
        int T  = wave + t2 * 8;
        int si = T >> 2, di = T & 3;
        v8f acc = {};
#pragma unroll
        for (int kk = 0; kk < 64; kk += 32) {
            v16bf a  = load_fragA(&ps[si * 16 + r][kk], g);
            v16bf bb = load_fragB(&vt[di * 16 + r][kk], g);
            acc = WMMA_BF16(a, bb, acc);
        }
#pragma unroll
        for (int j = 0; j < 8; j++) {
            int s = si * 16 + j + 8 * g;
            int d = di * 16 + r;
            attnout[((size_t)b * 64 + s) * DD + hh * 64 + d] = f2bf(acc[j]);
        }
    }
}

// ---------------------------------------------------------------------------
// Residual + LayerNorm: out = LN(x + res)*g + b; optional bf16 shadow copy.
// ---------------------------------------------------------------------------
__global__ __launch_bounds__(256) void ln_kernel(
    const float* __restrict__ x, const float* __restrict__ res,
    const float* __restrict__ gam, const float* __restrict__ bet,
    float* __restrict__ out, unsigned short* __restrict__ out_bf)
{
    const int row = blockIdx.x;
    const int tid = threadIdx.x;
    const float* xr = x + (size_t)row * DD;
    float v[4];
    float s = 0.f, s2 = 0.f;
#pragma unroll
    for (int i = 0; i < 4; i++) {
        int d = tid + i * 256;
        float t = xr[d];
        if (res) t += res[(size_t)row * DD + d];
        v[i] = t; s += t; s2 += t * t;
    }
#pragma unroll
    for (int o = 16; o > 0; o >>= 1) {
        s  += __shfl_xor(s, o, 32);
        s2 += __shfl_xor(s2, o, 32);
    }
    __shared__ float ws1[8], ws2[8];
    int wv = tid >> 5, ln = tid & 31;
    if (ln == 0) { ws1[wv] = s; ws2[wv] = s2; }
    __syncthreads();
    if (tid == 0) {
        float a = 0.f, bsum = 0.f;
        for (int i = 0; i < 8; i++) { a += ws1[i]; bsum += ws2[i]; }
        ws1[0] = a; ws2[0] = bsum;
    }
    __syncthreads();
    const float mean = ws1[0] * (1.0f / DD);
    const float var  = ws2[0] * (1.0f / DD) - mean * mean;
    const float rstd = rsqrtf(var + 1e-5f);
    float* orow = out + (size_t)row * DD;
#pragma unroll
    for (int i = 0; i < 4; i++) {
        int d = tid + i * 256;
        float y = (v[i] - mean) * rstd * gam[d] + bet[d];
        orow[d] = y;
        if (out_bf) out_bf[(size_t)row * DD + d] = f2bf(y);
    }
}

// ---------------------------------------------------------------------------
// Embedding: h = token_embed[x] + step_embed[min(step,63)]  (f32 + bf16 copy)
// ---------------------------------------------------------------------------
__global__ __launch_bounds__(256) void embed_kernel(
    const int* __restrict__ x, const int* __restrict__ step,
    const float* __restrict__ tok, const float* __restrict__ se,
    float* __restrict__ h, unsigned short* __restrict__ hb)
{
    size_t idx = (size_t)blockIdx.x * 256 + threadIdx.x;
    int row = (int)(idx >> 10);
    int d   = (int)(idx & 1023);
    int t   = x[row];
    int si  = step[0];
    if (si > 63) si = 63;
    if (si < 0)  si = 0;
    float y = tok[(size_t)t * DD + d] + se[(size_t)si * DD + d];
    h[idx]  = y;
    hb[idx] = f2bf(y);
}

// ---------------------------------------------------------------------------
// Mean pool over S (f32 + bf16 copy)
// ---------------------------------------------------------------------------
__global__ __launch_bounds__(256) void pool_kernel(
    const float* __restrict__ hf, float* __restrict__ pooled,
    unsigned short* __restrict__ pooledb)
{
    int b = blockIdx.x, tid = threadIdx.x;
#pragma unroll
    for (int i = 0; i < 4; i++) {
        int d = tid + i * 256;
        float s = 0.f;
        for (int ss = 0; ss < 64; ss++)
            s += hf[((size_t)b * 64 + ss) * DD + d];
        float y = s * (1.0f / 64.0f);
        pooled[(size_t)b * DD + d]  = y;
        pooledb[(size_t)b * DD + d] = f2bf(y);
    }
}

// ---------------------------------------------------------------------------
// Value head: out[b] = dot(pooled[b], val_w) + val_b
// ---------------------------------------------------------------------------
__global__ __launch_bounds__(256) void value_kernel(
    const float* __restrict__ pooled, const float* __restrict__ vw,
    const float* __restrict__ vb, float* __restrict__ out)
{
    int b = blockIdx.x, tid = threadIdx.x;
    float s = 0.f;
#pragma unroll
    for (int i = 0; i < 4; i++) {
        int d = tid + i * 256;
        s += pooled[(size_t)b * DD + d] * vw[d];
    }
#pragma unroll
    for (int o = 16; o > 0; o >>= 1) s += __shfl_xor(s, o, 32);
    __shared__ float ws[8];
    int wv = tid >> 5, ln = tid & 31;
    if (ln == 0) ws[wv] = s;
    __syncthreads();
    if (tid == 0) {
        float a = 0.f;
        for (int i = 0; i < 8; i++) a += ws[i];
        out[b] = a + vb[0];
    }
}

// ---------------------------------------------------------------------------
// Host-side orchestration
// ---------------------------------------------------------------------------
extern "C" void kernel_launch(void* const* d_in, const int* in_sizes, int n_in,
                              void* d_out, int out_size, void* d_ws, size_t ws_size,
                              hipStream_t stream)
{
    (void)in_sizes; (void)n_in; (void)out_size; (void)ws_size;

    const int*   x       = (const int*)  d_in[0];
    const int*   step    = (const int*)  d_in[1];
    const float* tok     = (const float*)d_in[2];
    const float* se      = (const float*)d_in[3];
    const float* qkv_w   = (const float*)d_in[4];
    const float* qkv_b   = (const float*)d_in[5];
    const float* out_w   = (const float*)d_in[6];
    const float* out_b   = (const float*)d_in[7];
    const float* rel     = (const float*)d_in[8];
    const float* w1      = (const float*)d_in[9];
    const float* b1      = (const float*)d_in[10];
    const float* w2      = (const float*)d_in[11];
    const float* b2      = (const float*)d_in[12];
    const float* ln1_g   = (const float*)d_in[13];
    const float* ln1_b   = (const float*)d_in[14];
    const float* ln2_g   = (const float*)d_in[15];
    const float* ln2_b   = (const float*)d_in[16];
    const float* lno_g   = (const float*)d_in[17];
    const float* lno_b   = (const float*)d_in[18];
    const float* pol_w   = (const float*)d_in[19];
    const float* pol_b   = (const float*)d_in[20];
    const float* val_w   = (const float*)d_in[21];
    const float* val_b   = (const float*)d_in[22];

    // workspace carve
    char* w = (char*)d_ws;
    float*          h       = (float*)w;          w += (size_t)TOK * DD * 4;
    unsigned short* hb      = (unsigned short*)w; w += (size_t)TOK * DD * 2;
    unsigned short* qkv     = (unsigned short*)w; w += (size_t)TOK * 3 * DD * 2;
    unsigned short* attn    = (unsigned short*)w; w += (size_t)TOK * DD * 2;
    float*          proj    = (float*)w;          w += (size_t)TOK * DD * 4;
    unsigned short* ffn1    = (unsigned short*)w; w += (size_t)TOK * DFF * 2;
    float*          pooled  = (float*)w;          w += (size_t)BB * DD * 4;
    unsigned short* pooledb = (unsigned short*)w; w += (size_t)BB * DD * 2;
    unsigned short* wqb     = (unsigned short*)w; w += (size_t)LL * 3 * DD * DD * 2;
    unsigned short* wob     = (unsigned short*)w; w += (size_t)LL * DD * DD * 2;
    unsigned short* w1b     = (unsigned short*)w; w += (size_t)LL * DFF * DD * 2;
    unsigned short* w2b     = (unsigned short*)w; w += (size_t)LL * DD * DFF * 2;
    unsigned short* polb    = (unsigned short*)w; w += (size_t)VV * DD * 2;

    float* out_policy = (float*)d_out;
    float* out_value  = out_policy + (size_t)BB * VV;
    float* out_h      = out_value + BB;

    // one-time (per launch) weight conversions fp32 -> bf16
    cvt_bf16_kernel<<<(LL * 3 * DD * DD) / 2048, 256, 0, stream>>>(qkv_w, wqb);
    cvt_bf16_kernel<<<(LL * DD * DD) / 2048,     256, 0, stream>>>(out_w, wob);
    cvt_bf16_kernel<<<(LL * DFF * DD) / 2048,    256, 0, stream>>>(w1, w1b);
    cvt_bf16_kernel<<<(LL * DD * DFF) / 2048,    256, 0, stream>>>(w2, w2b);
    cvt_bf16_kernel<<<(VV * DD) / 2048,          256, 0, stream>>>(pol_w, polb);

    embed_kernel<<<(TOK * DD) / 256, 256, 0, stream>>>(x, step, tok, se, h, hb);

    for (int l = 0; l < LL; l++) {
        // QKV: bf16 [TOK,1024] x bf16 [3072,1024]^T -> bf16 [TOK,3072]
        gemm_wmma_async<true, false><<<dim3(3 * DD / 64, TOK / 64), 256, 0, stream>>>(
            hb, wqb + (size_t)l * 3 * DD * DD, qkv_b + (size_t)l * 3 * DD, qkv, DD, 3 * DD);

        // fused attention
        attn_kernel<<<BB * HH, 256, 0, stream>>>(qkv, rel + (size_t)l * 127, attn);

        // out-projection -> fp32
        gemm_wmma_async<false, false><<<dim3(DD / 64, TOK / 64), 256, 0, stream>>>(
            attn, wob + (size_t)l * DD * DD, out_b + (size_t)l * DD, proj, DD, DD);

        // h = LN(h + proj)  (+ bf16 shadow)
        ln_kernel<<<TOK, 256, 0, stream>>>(h, proj, ln1_g + l * DD, ln1_b + l * DD, h, hb);

        // FFN1 + GELU -> bf16 [TOK,4096]
        gemm_wmma_async<true, true><<<dim3(DFF / 64, TOK / 64), 256, 0, stream>>>(
            hb, w1b + (size_t)l * DFF * DD, b1 + (size_t)l * DFF, ffn1, DD, DFF);

        // FFN2 -> fp32
        gemm_wmma_async<false, false><<<dim3(DD / 64, TOK / 64), 256, 0, stream>>>(
            ffn1, w2b + (size_t)l * DD * DFF, b2 + (size_t)l * DD, proj, DFF, DD);

        // h = LN(h + proj)  (+ bf16 shadow)
        ln_kernel<<<TOK, 256, 0, stream>>>(h, proj, ln2_g + l * DD, ln2_b + l * DD, h, hb);
    }

    // final LN -> h output region of d_out (no bf16 shadow needed)
    ln_kernel<<<TOK, 256, 0, stream>>>(h, nullptr, lno_g, lno_b, out_h, nullptr);

    // mean pool, heads
    pool_kernel<<<BB, 256, 0, stream>>>(out_h, pooled, pooledb);
    gemm_wmma_async<false, false><<<dim3(VV / 64, BB / 64), 256, 0, stream>>>(
        pooledb, polb, pol_b, out_policy, DD, VV);
    value_kernel<<<BB, 256, 0, stream>>>(pooled, val_w, val_b, out_value);
}
